// HubertWithKmeans_15152644620914
// MI455X (gfx1250) — matrix operand order repaired and checked by reading, not compiled, not run on hardware
//
#include <hip/hip_runtime.h>
#include <stdint.h>

// Problem constants (from reference): B=8, T=4096, D=768, C=2048
#define BT   32768          // B*T tokens
#define DIM  768            // feature dim (K)
#define NC   2048           // number of centers (N)
#define KCH  (DIM / 32)     // 24 K-chunks of 32 for V_WMMA_F32_16X16X32_BF16

// Plain (trivially-constructible) vector types, safe inside unions.
typedef __attribute__((ext_vector_type(16))) __bf16   v16bf;
typedef __attribute__((ext_vector_type(8)))  float    v8f;
typedef __attribute__((ext_vector_type(4)))  uint32_t u32x4;
typedef __attribute__((ext_vector_type(4)))  float    f32x4;

union Frag {
  v16bf          v;
  unsigned short u[16];
  u32x4          q[2];
};

__device__ __forceinline__ unsigned short f32_to_bf16_rne(float f) {
  uint32_t u = __float_as_uint(f);
  u += 0x7FFFu + ((u >> 16) & 1u);       // round-to-nearest-even
  return (unsigned short)(u >> 16);
}
__device__ __forceinline__ float bf16_us_to_f32(unsigned short s) {
  return __uint_as_float(((uint32_t)s) << 16);
}

// WGP-scope prefetch (scope field 0 = WGP: pulls into ALL cache levels,
// including WGP$). __builtin_prefetch can only reach SE/SYS scope, which
// skips the near cache (ISA §10.5) -- useless here since centers are
// already L2-resident after the first sweep.
__device__ __forceinline__ void prefetch_wgp(const void* p) {
  asm volatile("global_prefetch_b8 %0, off"
               :: "v"((unsigned long long)(uintptr_t)p));
}

// ---------------------------------------------------------------------------
// Prep: centers (fp32, NC x DIM) -> bf16 hi/lo split arrays + ||c||^2 (fp32)
// ---------------------------------------------------------------------------
__global__ __launch_bounds__(256) void prep_centers_kernel(
    const float* __restrict__ centers,
    unsigned short* __restrict__ chi,
    unsigned short* __restrict__ clo,
    float* __restrict__ csq) {
  __shared__ float red[256];
  const int n = blockIdx.x;        // one center per block
  const int t = threadIdx.x;
  const float* row = centers + (size_t)n * DIM;
  float acc = 0.f;
  for (int k = t; k < DIM; k += 256) {
    float x = row[k];
    unsigned short h = f32_to_bf16_rne(x);
    unsigned short l = f32_to_bf16_rne(x - bf16_us_to_f32(h));
    chi[(size_t)n * DIM + k] = h;
    clo[(size_t)n * DIM + k] = l;
    acc += x * x;                  // full fp32 for the ||c||^2 term
  }
  red[t] = acc;
  __syncthreads();
  for (int s = 128; s > 0; s >>= 1) {
    if (t < s) red[t] += red[t + s];
    __syncthreads();
  }
  if (t == 0) csq[n] = red[0];
}

// ---------------------------------------------------------------------------
// Main: each wave owns 16 token rows (A register-resident, bf16 hi/lo split),
// sweeps all 2048 centers in 16-wide tiles with 3 chained WMMA accumulators,
// double-buffered B fragment loads, and a running per-lane (min, argmin).
// ---------------------------------------------------------------------------
__global__ __launch_bounds__(256, 2) void kmeans_assign_kernel(
    const float* __restrict__ embed,
    const unsigned short* __restrict__ chi,
    const unsigned short* __restrict__ clo,
    const float* __restrict__ csq,
    float* __restrict__ out) {
  const int lane  = threadIdx.x & 31;
  const int wave  = threadIdx.x >> 5;
  const int m0    = blockIdx.x * 128 + wave * 16;  // first token row of this wave
  const int laneM = lane & 15;                     // A: M index / B: N index
  const int half  = lane >> 4;                     // lane group (0/1)
  const int kbase = half * 16;                     // A fragment K sub-base

  // ---- Load A (16 rows x 768 fp32) once; convert to bf16 hi/lo fragments.
  // A 16x32 bf16 layout: lane L holds K = kbase..kbase+15 (contiguous)
  // of row M = L%16, packed 2 per VGPR, low K in [15:0].
  v16bf Ahi[KCH], Alo[KCH];
  {
    const float* rowp = embed + (size_t)(m0 + laneM) * DIM + kbase;
#pragma unroll
    for (int c = 0; c < KCH; ++c) {
      const float* p = rowp + c * 32;
      f32x4 a0 = *(const f32x4*)(p + 0);
      f32x4 a1 = *(const f32x4*)(p + 4);
      f32x4 a2 = *(const f32x4*)(p + 8);
      f32x4 a3 = *(const f32x4*)(p + 12);
      float f[16] = {a0.x, a0.y, a0.z, a0.w, a1.x, a1.y, a1.z, a1.w,
                     a2.x, a2.y, a2.z, a2.w, a3.x, a3.y, a3.z, a3.w};
      Frag h, l;
#pragma unroll
      for (int e = 0; e < 16; ++e) {
        unsigned short hb = f32_to_bf16_rne(f[e]);
        h.u[e] = hb;
        l.u[e] = f32_to_bf16_rne(f[e] - bf16_us_to_f32(hb));
      }
      Ahi[c] = h.v;
      Alo[c] = l.v;
    }
  }

  float minval[8];
  int   minidx[8];
#pragma unroll
  for (int r = 0; r < 8; ++r) { minval[r] = 3.4e38f; minidx[r] = 0; }

  // B 32x16 bf16 layout: lane L holds K = (L/16)*16 + 0..15 (contiguous) of
  // center N = L%16  -> 32 contiguous bytes = two b128 loads per fragment.
  const size_t bcol = (size_t)laneM * DIM + (size_t)half * 16;

  for (int nt = 0; nt < NC / 16; ++nt) {
    const int n = nt * 16 + laneM;                       // this lane's column
    const unsigned short* bph = chi + (size_t)nt * 16 * DIM + bcol;
    const unsigned short* bpl = clo + (size_t)nt * 16 * DIM + bcol;

    // Warm the WGP$ one N-tile ahead.
    if (nt + 1 < NC / 16) {
      prefetch_wgp(bph + (size_t)16 * DIM);
      prefetch_wgp(bpl + (size_t)16 * DIM);
    }
    const float cs = csq[n];                             // hoisted above WMMAs

    v8f acc0 = {}, acc1 = {}, acc2 = {};                 // 3 chains for ILP

    // Double-buffered B fragments: load chunk c+1 before consuming chunk c,
    // so each s_wait_loadcnt has a full chunk of WMMA work behind it.
    Frag bh[2], bl[2];
    {
      const u32x4* qh = (const u32x4*)(bph);
      bh[0].q[0] = qh[0]; bh[0].q[1] = qh[1];
      const u32x4* ql = (const u32x4*)(bpl);
      bl[0].q[0] = ql[0]; bl[0].q[1] = ql[1];
    }
#pragma unroll
    for (int c = 0; c < KCH; ++c) {
      const int cur = c & 1;
      const int nxt = cur ^ 1;
      if (c + 1 < KCH) {
        const u32x4* qh = (const u32x4*)(bph + (c + 1) * 32);
        bh[nxt].q[0] = qh[0]; bh[nxt].q[1] = qh[1];
        const u32x4* ql = (const u32x4*)(bpl + (c + 1) * 32);
        bl[nxt].q[0] = ql[0]; bl[nxt].q[1] = ql[1];
      }
      // split-GEMM: dot ~= hi*hi + hi*lo + lo*hi  (fp32 accumulate)
      acc0 = __builtin_amdgcn_wmma_f32_16x16x32_bf16(false, Ahi[c], false, bh[cur].v,
                                                     (short)0, acc0, false, false);
      acc1 = __builtin_amdgcn_wmma_f32_16x16x32_bf16(false, Ahi[c], false, bl[cur].v,
                                                     (short)0, acc1, false, false);
      acc2 = __builtin_amdgcn_wmma_f32_16x16x32_bf16(false, Alo[c], false, bh[cur].v,
                                                     (short)0, acc2, false, false);
    }

    // D layout: lane L, VGPR r -> (M = r + 8*(L/16), N = L%16)
#pragma unroll
    for (int r = 0; r < 8; ++r) {
      float dot = acc0[r] + acc1[r] + acc2[r];
      float s = cs - 2.0f * dot;                 // dist^2 - ||x||^2 (monotone)
      if (s < minval[r]) { minval[r] = s; minidx[r] = n; }
    }
  }

  // Reduce (val, idx) across the 16 lanes of each half; lower index wins ties.
#pragma unroll
  for (int r = 0; r < 8; ++r) {
    float v = minval[r];
    int   i = minidx[r];
#pragma unroll
    for (int off = 8; off >= 1; off >>= 1) {
      float ov = __shfl_xor(v, off, 32);
      int   oi = __shfl_xor(i, off, 32);
      if (ov < v || (ov == v && oi < i)) { v = ov; i = oi; }
    }
    if (laneM == 0) out[m0 + r + 8 * half] = (float)i;
  }
}

// ---------------------------------------------------------------------------
extern "C" void kernel_launch(void* const* d_in, const int* in_sizes, int n_in,
                              void* d_out, int out_size, void* d_ws, size_t ws_size,
                              hipStream_t stream) {
  (void)in_sizes; (void)n_in; (void)out_size; (void)ws_size;
  const float* embed   = (const float*)d_in[0];   // (8,4096,768) fp32
  const float* centers = (const float*)d_in[1];   // (2048,768)  fp32

  // Workspace layout: chi (3.07 MB) | clo (3.07 MB) | csq (8 KB)  ~ 6.3 MB
  unsigned short* chi = (unsigned short*)d_ws;
  unsigned short* clo = chi + (size_t)NC * DIM;
  float*          csq = (float*)(clo + (size_t)NC * DIM);
  float*          out = (float*)d_out;            // 32768 cluster ids

  prep_centers_kernel<<<NC, 256, 0, stream>>>(centers, chi, clo, csq);
  kmeans_assign_kernel<<<BT / 128, 256, 0, stream>>>(embed, chi, clo, csq, out);
}
